// GPT_2181843386952
// MI455X (gfx1250) — matrix operand order repaired and checked by reading, not compiled, not run on hardware
//
#include <hip/hip_runtime.h>
#include <hip/hip_bf16.h>

// ---------------------------------------------------------------------------
// BigBird-GPT forward for MI455X (gfx1250, wave32, WMMA).
// All GEMMs use v_wmma_f32_16x16x32_bf16 (bf16 in, f32 accumulate).
// V projection is stored TRANSPOSED (vT[b][d][s]) so attention's P@V
// B-fragments are contiguous b128 LDS loads (no scalar transpose gather).
// Workspace requirement: ~126 MB.
// ---------------------------------------------------------------------------

#define S_LEN   1024
#define D_MODEL 768
#define N_HEAD  12
#define D_HEAD  64
#define BATCH   8
#define VOCAB   512
#define N_LAYER 8
#define M_ROWS  (BATCH * S_LEN)   // 8192
#define D_FF    3072
#define NB_WIN  32
#define NM_GLOB 64

typedef __attribute__((ext_vector_type(16))) __bf16          v16bf;
typedef __attribute__((ext_vector_type(8)))  float           v8f;
typedef __attribute__((ext_vector_type(8)))  unsigned short  u16x8;

static __device__ __forceinline__ unsigned short f32_to_bf16_rne(float f) {
  union { float f; unsigned u; } x; x.f = f;
  unsigned r = x.u + 0x7FFFu + ((x.u >> 16) & 1u);
  return (unsigned short)(r >> 16);
}

// Build a 16-element bf16 fragment from two 16B-aligned 8-element runs.
static __device__ __forceinline__ v16bf make_frag(const unsigned short* p0,
                                                  const unsigned short* p1) {
  union { struct { u16x8 lo, hi; } s; v16bf f; } u;
  u.s.lo = *(const u16x8*)p0;
  u.s.hi = *(const u16x8*)p1;
  return u.f;
}

// ---------------------------------------------------------------------------
// Embedding: h[b,s,:] = wte[x[b,s],:] + pos[s,:]
// ---------------------------------------------------------------------------
__global__ __launch_bounds__(256) void embed_kernel(
    const int* __restrict__ x, const float* __restrict__ wte,
    const float* __restrict__ pos, float* __restrict__ h)
{
  const int rs  = blockIdx.x;
  const int s   = rs & (S_LEN - 1);
  const int tok = x[rs];
  #pragma unroll
  for (int i = 0; i < 3; ++i) {
    const int d = threadIdx.x + 256 * i;
    h[(size_t)rs * D_MODEL + d] = wte[(size_t)tok * D_MODEL + d] + pos[(size_t)s * D_MODEL + d];
  }
}

// ---------------------------------------------------------------------------
// LayerNorm (f32 in) -> bf16 out, one row per block.
// ---------------------------------------------------------------------------
__global__ __launch_bounds__(256) void layernorm_to_bf16(
    const float* __restrict__ x, const float* __restrict__ g,
    const float* __restrict__ b, unsigned short* __restrict__ out)
{
  const int row = blockIdx.x;
  const int tid = threadIdx.x;
  __shared__ float red[256];
  const float* xr = x + (size_t)row * D_MODEL;
  float v[3];
  float s = 0.f, s2 = 0.f;
  #pragma unroll
  for (int i = 0; i < 3; ++i) {
    v[i] = xr[tid + 256 * i];
    s  += v[i];
    s2 += v[i] * v[i];
  }
  red[tid] = s; __syncthreads();
  for (int off = 128; off > 0; off >>= 1) { if (tid < off) red[tid] += red[tid + off]; __syncthreads(); }
  const float mean = red[0] * (1.0f / D_MODEL);
  __syncthreads();
  red[tid] = s2; __syncthreads();
  for (int off = 128; off > 0; off >>= 1) { if (tid < off) red[tid] += red[tid + off]; __syncthreads(); }
  const float var  = red[0] * (1.0f / D_MODEL) - mean * mean;
  const float rstd = rsqrtf(var + 1e-5f);
  #pragma unroll
  for (int i = 0; i < 3; ++i) {
    const int d = tid + 256 * i;
    out[(size_t)row * D_MODEL + d] = f32_to_bf16_rne((v[i] - mean) * rstd * g[d] + b[d]);
  }
}

// ---------------------------------------------------------------------------
// WMMA GEMM: C[M,N] = A[M,K](bf16) @ W[K,N](f32, cast to bf16) (+ bias)
// Block tile 128x128, 8 waves, each wave 64x32 (4x2 WMMA tiles), k-step 32.
// MODE 0: store bf16(acc+bias)      MODE 1: store bf16(gelu(acc+bias))
// MODE 2: f32 out += acc+bias       MODE 3: f32 out = acc   (no bias)
// MODE 4: store bf16(acc+bias) transposed as vT[b][n][s]  (for V projection)
// ---------------------------------------------------------------------------
template <int MODE>
__global__ __launch_bounds__(256) void gemm_bf16_wmma(
    const unsigned short* __restrict__ A, const float* __restrict__ W,
    const float* __restrict__ bias, void* __restrict__ out,
    int M, int N, int K)
{
  __shared__ alignas(16) unsigned short As[128][40];  // [m][k], 80B stride
  __shared__ alignas(16) unsigned short Bs[128][40];  // [n][k] (transposed)
  const int tid  = threadIdx.x;
  const int lane = tid & 31;
  const int wid  = tid >> 5;
  const int lh   = lane >> 4;
  const int l15  = lane & 15;
  const int m0 = blockIdx.y * 128;
  const int n0 = blockIdx.x * 128;
  const int wm = (wid & 1) * 64;
  const int wn = (wid >> 1) * 32;

  v8f acc[4][2] = {};

  const int ar = tid >> 1;
  const int ak = (tid & 1) * 16;
  const int bk2 = (tid >> 4) * 2;       // 0,2,...,30
  const int bn  = (tid & 15) * 8;       // 0,8,...,120

  for (int k0 = 0; k0 < K; k0 += 32) {
    __syncthreads();
    { // stage A: 16 bf16 = two b128 loads + two b128 LDS stores
      const u16x8* src = (const u16x8*)(A + (size_t)(m0 + ar) * K + k0 + ak);
      *(u16x8*)&As[ar][ak]     = src[0];
      *(u16x8*)&As[ar][ak + 8] = src[1];
    }
    { // stage W: two rows of 8 consecutive f32 (float4-coalesced),
      // pack k-pairs -> one b32 LDS store per n
      const float* r0 = W + (size_t)(k0 + bk2) * N + n0 + bn;
      const float* r1 = r0 + N;
      float4 a0 = ((const float4*)r0)[0], a1 = ((const float4*)r0)[1];
      float4 b0 = ((const float4*)r1)[0], b1 = ((const float4*)r1)[1];
      float ra[8] = {a0.x, a0.y, a0.z, a0.w, a1.x, a1.y, a1.z, a1.w};
      float rb[8] = {b0.x, b0.y, b0.z, b0.w, b1.x, b1.y, b1.z, b1.w};
      #pragma unroll
      for (int i = 0; i < 8; ++i) {
        const unsigned pk = (unsigned)f32_to_bf16_rne(ra[i]) |
                            ((unsigned)f32_to_bf16_rne(rb[i]) << 16);
        *(unsigned*)&Bs[bn + i][bk2] = pk;
      }
      if (k0 + 32 < K)
        __builtin_prefetch(W + (size_t)(k0 + 32 + bk2) * N + n0 + bn, 0, 1);
    }
    __syncthreads();

    v16bf af[4];
    #pragma unroll
    for (int i = 0; i < 4; ++i) {
      const int row  = wm + 16 * i + l15;
      const int koff = lh * 8;
      af[i] = make_frag(&As[row][koff], &As[row][16 + koff]);
    }
    v16bf bfg[2];
    #pragma unroll
    for (int j = 0; j < 2; ++j) {
      const int ncol = wn + 16 * j + l15;
      bfg[j] = make_frag(&Bs[ncol][8 * lh], &Bs[ncol][16 + 8 * lh]);
    }
    #pragma unroll
    for (int i = 0; i < 4; ++i)
      #pragma unroll
      for (int j = 0; j < 2; ++j)
        acc[i][j] = __builtin_amdgcn_wmma_f32_16x16x32_bf16(
            false, af[i], false, bfg[j], (short)0, acc[i][j], false, false);
  }

  #pragma unroll
  for (int i = 0; i < 4; ++i) {
    #pragma unroll
    for (int j = 0; j < 2; ++j) {
      #pragma unroll
      for (int e = 0; e < 8; ++e) {
        const int m = m0 + wm + 16 * i + e + 8 * lh;
        const int n = n0 + wn + 16 * j + l15;
        float val = acc[i][j][e];
        if (MODE != 3) val += bias[n];
        if (MODE == 0) {
          ((unsigned short*)out)[(size_t)m * N + n] = f32_to_bf16_rne(val);
        } else if (MODE == 1) {
          const float ge = 0.5f * val * (1.0f + erff(val * 0.70710678118654752f));
          ((unsigned short*)out)[(size_t)m * N + n] = f32_to_bf16_rne(ge);
        } else if (MODE == 2) {
          ((float*)out)[(size_t)m * N + n] += val;
        } else if (MODE == 3) {
          ((float*)out)[(size_t)m * N + n] = val;
        } else { // MODE 4: transposed store vT[b][n][s], m = b*S + s
          const int bb = m >> 10;           // m / S_LEN
          const int ss = m & (S_LEN - 1);   // m % S_LEN
          ((unsigned short*)out)[((size_t)bb * D_MODEL + n) * S_LEN + ss] =
              f32_to_bf16_rne(val);
        }
      }
    }
  }
}

// ---------------------------------------------------------------------------
// BigBird attention, flash-style, one wave per (16-query tile, head, batch).
// q,k: [b][s][d] bf16.  vT: [b][d][s] bf16 (pre-transposed by the V GEMM).
// ---------------------------------------------------------------------------
__global__ __launch_bounds__(32) void bigbird_attn(
    const unsigned short* __restrict__ qb, const unsigned short* __restrict__ kb,
    const unsigned short* __restrict__ vT, float* __restrict__ h)
{
  const int qt  = blockIdx.x;
  const int hd  = blockIdx.y;
  const int b   = blockIdx.z;
  const int lane = threadIdx.x;
  const int lh   = lane >> 4;
  const int l15  = lane & 15;
  const int q0   = qt * 16;

  __shared__ alignas(16) unsigned short qs[16][72];   // [q][d],  144B stride
  __shared__ alignas(16) unsigned short ks[32][72];   // [key][d]
  __shared__ alignas(16) unsigned short vsT[64][40];  // [d][key], 80B stride
  __shared__ alignas(16) unsigned short ps[16][40];   // [q][key]

  const size_t base  = ((size_t)b * S_LEN) * D_MODEL + (size_t)hd * D_HEAD;
  const size_t baseT = ((size_t)b * D_MODEL + (size_t)hd * D_HEAD) * S_LEN;

  { // load Q tile 16x64: b128 moves
    const int r  = lane >> 1;
    const int c0 = (lane & 1) * 32;
    const u16x8* src = (const u16x8*)(qb + base + (size_t)(q0 + r) * D_MODEL + c0);
    #pragma unroll
    for (int i = 0; i < 4; ++i) *(u16x8*)&qs[r][c0 + 8 * i] = src[i];
  }
  __syncthreads();

  v16bf aq[2];
  #pragma unroll
  for (int kc = 0; kc < 2; ++kc) {
    const int koff = lh * 8 + 32 * kc;
    aq[kc] = make_frag(&qs[l15][koff], &qs[l15][16 + koff]);
  }

  v8f o[4] = {};
  float m8[8], l8[8];
  #pragma unroll
  for (int e = 0; e < 8; ++e) { m8[e] = -3.0e38f; l8[e] = 0.f; }

  const float scale = 0.125f;  // 64^-0.5
  const int cw1 = (q0 + 15) / 32;
  const int cw0 = (q0 >= NB_WIN) ? (q0 - NB_WIN) / 32 : 0;

  for (int c = 0; c <= cw1; ++c) {
    if (c > 1 && c < cw0) continue;  // neither global (keys<64) nor in window
    const int k0 = c * 32;
    __syncthreads();
    { // K chunk rows [key][d]: b128 moves (row per lane)
      const u16x8* ksrc = (const u16x8*)(kb + base + (size_t)(k0 + lane) * D_MODEL);
      #pragma unroll
      for (int i = 0; i < 8; ++i) *(u16x8*)&ks[lane][8 * i] = ksrc[i];
    }
    { // V chunk from vT [d][key]: contiguous keys per d row, b128 moves
      const u16x8* v0 = (const u16x8*)(vT + baseT + (size_t)lane * S_LEN + k0);
      const u16x8* v1 = (const u16x8*)(vT + baseT + (size_t)(lane + 32) * S_LEN + k0);
      #pragma unroll
      for (int i = 0; i < 4; ++i) {
        *(u16x8*)&vsT[lane][8 * i]      = v0[i];
        *(u16x8*)&vsT[lane + 32][8 * i] = v1[i];
      }
    }
    __syncthreads();

    // scores: two 16-key subtiles, K=64 contraction = 2 WMMAs each
    v8f sc[2] = {};
    #pragma unroll
    for (int st = 0; st < 2; ++st) {
      const int key = 16 * st + l15;
      #pragma unroll
      for (int kc = 0; kc < 2; ++kc) {
        const int dbase = 8 * lh + 32 * kc;
        const v16bf bk = make_frag(&ks[key][dbase], &ks[key][dbase + 16]);
        sc[st] = __builtin_amdgcn_wmma_f32_16x16x32_bf16(
            false, aq[kc], false, bk, (short)0, sc[st], false, false);
      }
    }

    // mask + online softmax (row r of tile lives in VGPR e=r%8, lane-half r/8)
    float p0a[8], p1a[8];
    #pragma unroll
    for (int e = 0; e < 8; ++e) {
      const int row  = q0 + e + 8 * lh;
      const int key0 = k0 + l15;
      const int key1 = key0 + 16;
      float v0 = sc[0][e] * scale;
      float v1 = sc[1][e] * scale;
      const bool ok0 = (key0 <= row) && ((row - key0) <= NB_WIN || key0 < NM_GLOB);
      const bool ok1 = (key1 <= row) && ((row - key1) <= NB_WIN || key1 < NM_GLOB);
      v0 = ok0 ? v0 : -1.0e30f;
      v1 = ok1 ? v1 : -1.0e30f;
      float mv = fmaxf(v0, v1);
      #pragma unroll
      for (int off = 1; off < 16; off <<= 1) mv = fmaxf(mv, __shfl_xor(mv, off, 32));
      const float mnew = fmaxf(m8[e], mv);
      const float corr = expf(m8[e] - mnew);
      const float p0 = expf(v0 - mnew);
      const float p1 = expf(v1 - mnew);
      float rs = p0 + p1;
      #pragma unroll
      for (int off = 1; off < 16; off <<= 1) rs += __shfl_xor(rs, off, 32);
      l8[e] = l8[e] * corr + rs;
      m8[e] = mnew;
      p0a[e] = p0; p1a[e] = p1;
      #pragma unroll
      for (int n = 0; n < 4; ++n) o[n][e] *= corr;
    }

    // P (16x32) -> LDS -> A-frag, then O += P @ V via 4 WMMAs
    __syncthreads();
    #pragma unroll
    for (int e = 0; e < 8; ++e) {
      const int r = e + 8 * lh;
      ps[r][l15]      = f32_to_bf16_rne(p0a[e]);
      ps[r][16 + l15] = f32_to_bf16_rne(p1a[e]);
    }
    __syncthreads();
    const int koff = lh * 8;
    const v16bf ap = make_frag(&ps[l15][koff], &ps[l15][16 + koff]);
    #pragma unroll
    for (int n = 0; n < 4; ++n) {
      const int ncol = 16 * n + l15;
      const v16bf bvf = make_frag(&vsT[ncol][8 * lh], &vsT[ncol][16 + 8 * lh]);
      o[n] = __builtin_amdgcn_wmma_f32_16x16x32_bf16(
          false, ap, false, bvf, (short)0, o[n], false, false);
    }
  }

  // residual: h += O / l   (each (head, q-tile) owns a disjoint h slice)
  #pragma unroll
  for (int e = 0; e < 8; ++e) {
    const float inv = 1.0f / l8[e];
    const int row = q0 + e + 8 * lh;
    float* dst = h + ((size_t)b * S_LEN + row) * D_MODEL + hd * D_HEAD;
    #pragma unroll
    for (int n = 0; n < 4; ++n) dst[16 * n + l15] += o[n][e] * inv;
  }
}

// ---------------------------------------------------------------------------
// Loss: per-row NLL then mean reduction.
// ---------------------------------------------------------------------------
__global__ __launch_bounds__(256) void row_nll(
    const float* __restrict__ pred, const int* __restrict__ tgt,
    float* __restrict__ rowloss)
{
  const int row = blockIdx.x;
  const int tid = threadIdx.x;
  __shared__ float red[256];
  const float* p = pred + (size_t)row * VOCAB;
  const float a = p[tid], b = p[tid + 256];
  red[tid] = fmaxf(a, b); __syncthreads();
  for (int off = 128; off > 0; off >>= 1) { if (tid < off) red[tid] = fmaxf(red[tid], red[tid + off]); __syncthreads(); }
  const float m = red[0];
  __syncthreads();
  red[tid] = expf(a - m) + expf(b - m); __syncthreads();
  for (int off = 128; off > 0; off >>= 1) { if (tid < off) red[tid] += red[tid + off]; __syncthreads(); }
  if (tid == 0) rowloss[row] = (m + logf(red[0])) - p[tgt[row]];
}

__global__ __launch_bounds__(256) void loss_reduce(
    const float* __restrict__ rowloss, float* __restrict__ out)
{
  const int tid = threadIdx.x;
  __shared__ float red[256];
  float s = 0.f;
  for (int i = tid; i < M_ROWS; i += 256) s += rowloss[i];
  red[tid] = s; __syncthreads();
  for (int off = 128; off > 0; off >>= 1) { if (tid < off) red[tid] += red[tid + off]; __syncthreads(); }
  if (tid == 0) out[0] = red[0] * (1.0f / M_ROWS);
}

// ---------------------------------------------------------------------------
extern "C" void kernel_launch(void* const* d_in, const int* in_sizes, int n_in,
                              void* d_out, int out_size, void* d_ws, size_t ws_size,
                              hipStream_t stream)
{
  (void)in_sizes; (void)n_in; (void)out_size; (void)ws_size;
  const int*   x       = (const int*)  d_in[0];
  const int*   targets = (const int*)  d_in[1];
  const float* wte     = (const float*)d_in[2];
  const float* pos     = (const float*)d_in[3];
  const float* ln1_g   = (const float*)d_in[4];
  const float* ln1_b   = (const float*)d_in[5];
  const float* wq      = (const float*)d_in[6];
  const float* bq      = (const float*)d_in[7];
  const float* wk      = (const float*)d_in[8];
  const float* bk      = (const float*)d_in[9];
  const float* wv      = (const float*)d_in[10];
  const float* bv      = (const float*)d_in[11];
  const float* ln2_g   = (const float*)d_in[12];
  const float* ln2_b   = (const float*)d_in[13];
  const float* w1      = (const float*)d_in[14];
  const float* b1      = (const float*)d_in[15];
  const float* w2      = (const float*)d_in[16];
  const float* b2      = (const float*)d_in[17];
  const float* lnf_g   = (const float*)d_in[18];
  const float* lnf_b   = (const float*)d_in[19];
  const float* w_out   = (const float*)d_in[20];

  char* ws = (char*)d_ws;
  float*          h    = (float*)ws;          ws += (size_t)M_ROWS * D_MODEL * 4;
  unsigned short* abuf = (unsigned short*)ws; ws += (size_t)M_ROWS * D_MODEL * 2;
  unsigned short* qbuf = (unsigned short*)ws; ws += (size_t)M_ROWS * D_MODEL * 2;
  unsigned short* kbuf = (unsigned short*)ws; ws += (size_t)M_ROWS * D_MODEL * 2;
  unsigned short* vbuf = (unsigned short*)ws; ws += (size_t)M_ROWS * D_MODEL * 2; // vT layout
  unsigned short* gbuf = (unsigned short*)ws; ws += (size_t)M_ROWS * D_FF * 2;
  float*       rowloss = (float*)ws;          ws += (size_t)M_ROWS * 4;

  embed_kernel<<<M_ROWS, 256, 0, stream>>>(x, wte, pos, h);

  const dim3 g_qkv(D_MODEL / 128, M_ROWS / 128);
  const dim3 g_ff1(D_FF / 128,    M_ROWS / 128);
  const dim3 g_ff2(D_MODEL / 128, M_ROWS / 128);
  const dim3 g_att(S_LEN / 16, N_HEAD, BATCH);

  for (int l = 0; l < N_LAYER; ++l) {
    layernorm_to_bf16<<<M_ROWS, 256, 0, stream>>>(h, ln1_g + l * D_MODEL, ln1_b + l * D_MODEL, abuf);
    gemm_bf16_wmma<0><<<g_qkv, 256, 0, stream>>>(abuf, wq + (size_t)l * D_MODEL * D_MODEL,
                                                 bq + l * D_MODEL, qbuf, M_ROWS, D_MODEL, D_MODEL);
    gemm_bf16_wmma<0><<<g_qkv, 256, 0, stream>>>(abuf, wk + (size_t)l * D_MODEL * D_MODEL,
                                                 bk + l * D_MODEL, kbuf, M_ROWS, D_MODEL, D_MODEL);
    gemm_bf16_wmma<4><<<g_qkv, 256, 0, stream>>>(abuf, wv + (size_t)l * D_MODEL * D_MODEL,
                                                 bv + l * D_MODEL, vbuf, M_ROWS, D_MODEL, D_MODEL);
    bigbird_attn<<<g_att, 32, 0, stream>>>(qbuf, kbuf, vbuf, h);
    layernorm_to_bf16<<<M_ROWS, 256, 0, stream>>>(h, ln2_g + l * D_MODEL, ln2_b + l * D_MODEL, abuf);
    gemm_bf16_wmma<1><<<g_ff1, 256, 0, stream>>>(abuf, w1 + (size_t)l * D_MODEL * D_FF,
                                                 b1 + l * D_FF, gbuf, M_ROWS, D_FF, D_MODEL);
    gemm_bf16_wmma<2><<<g_ff2, 256, 0, stream>>>(gbuf, w2 + (size_t)l * D_FF * D_MODEL,
                                                 b2 + l * D_MODEL, h, M_ROWS, D_MODEL, D_FF);
  }

  layernorm_to_bf16<<<M_ROWS, 256, 0, stream>>>(h, lnf_g, lnf_b, abuf);
  float* pred = (float*)d_out;
  gemm_bf16_wmma<3><<<dim3(VOCAB / 128, M_ROWS / 128), 256, 0, stream>>>(
      abuf, w_out, nullptr, pred, M_ROWS, VOCAB, D_MODEL);

  row_nll<<<M_ROWS, 256, 0, stream>>>(pred, targets, rowloss);
  loss_reduce<<<1, 256, 0, stream>>>(rowloss, pred + (size_t)M_ROWS * VOCAB);
}
